// Model2_64020782514906
// MI455X (gfx1250) — matrix operand order repaired and checked by reading, not compiled, not run on hardware
//
#include <hip/hip_runtime.h>
#include <hip/hip_bf16.h>

// MI455X (gfx1250) implementation. Memory-bound workload (~28 GFLOP vs ~1 GB of
// traffic at 23.3 TB/s); all GEMM stages use v_wmma_f32_16x16x32_f16 (f32 accum),
// intermediates stored in f16 to halve HBM traffic. One (b,h) problem per block.

constexpr int B_   = 8;
constexpr int CIN  = 512;
constexpr int H_   = 112;
constexpr int W_   = 112;
constexpr int COUT = 128;   // channels of Q / S
constexpr int G_   = 4;
constexpr int CG   = CIN / G_;   // 128  (in-channels per group)
constexpr int OG   = COUT / G_;  // 32   (out-channels per group)
constexpr int OC   = 1025;       // output channels: 512 att + 512 flow + 1 V
constexpr int LDH  = 136;        // LDS stride in halves: 272B rows, 16B-aligned, bank-rotating
constexpr int NBH  = B_ * H_;    // 896 independent (b,h) problems

typedef __attribute__((ext_vector_type(16))) _Float16 v16h;
typedef __attribute__((ext_vector_type(8)))  float    v8f;

union FragH { v16h v; uint4 q[2]; };

// A operand (16x32 f16, M x K): row-major tile at p, row stride LDH.
// Lane layout per ISA: m = lane&15; lanes 0-15 hold K = k0+[0..7], k0+16+[0..7];
// lanes 16-31 hold K = k0+8+[0..7], k0+24+[0..7].
__device__ __forceinline__ v16h frag_A(const _Float16* p, int k0) {
  const int lane = threadIdx.x & 31;
  const int m = lane & 15, hh = lane >> 4;
  const _Float16* r = p + m * LDH + k0 + 8 * hh;
  FragH f;
  f.q[0] = *reinterpret_cast<const uint4*>(r);
  f.q[1] = *reinterpret_cast<const uint4*>(r + 16);
  return f.v;
}

// B operand (32x16 f16, K x N) supplied as B-transposed (N x K row-major) at p.
// Lane layout: n = lane&15; lanes 0-15 hold K = k0+[0..7], k0+16+[0..7];
// lanes 16-31 hold K = k0+8+[0..7], k0+24+[0..7].
__device__ __forceinline__ v16h frag_BT(const _Float16* p, int k0) {
  const int lane = threadIdx.x & 31;
  const int n = lane & 15, hh = lane >> 4;
  const _Float16* r = p + n * LDH + k0 + 8 * hh;
  FragH f;
  f.q[0] = *reinterpret_cast<const uint4*>(r);
  f.q[1] = *reinterpret_cast<const uint4*>(r + 16);
  return f.v;
}

// ---------------------------------------------------------------------------
// K1: grouped 1x1 conv.  Per (b,h,side): Q_g(32x112) = W_g(32x128) * Xg(128x112).
// Output stored transposed as [w][c] f16 slabs (that is exactly the A / B^T
// layout the score GEMM needs).
// ---------------------------------------------------------------------------
__global__ __launch_bounds__(256) void gconv_kernel(
    const float* __restrict__ xL, const float* __restrict__ xR,
    const float* __restrict__ wq, const float* __restrict__ bq,
    const float* __restrict__ wsW, const float* __restrict__ bsB,
    _Float16* __restrict__ Qt, _Float16* __restrict__ St) {
  __shared__ __align__(16) _Float16 sX[W_ * LDH];   // Xg^T  [w][i]
  __shared__ __align__(16) _Float16 sW[OG * LDH];   // Wg    [o][i]
  __shared__ float sBias[OG];

  const int bh = blockIdx.x;
  const int b = bh / H_, h = bh % H_;
  const int side = blockIdx.y;
  const float* x  = side ? xR  : xL;
  const float* wm = side ? wsW : wq;
  const float* bi = side ? bsB : bq;
  _Float16* outSlab = (side ? St : Qt) + (size_t)bh * (W_ * COUT);

  const int tid = threadIdx.x;
  const int wv = tid >> 5, lane = tid & 31, nl = lane & 15, hh = lane >> 4;

  for (int g = 0; g < G_; ++g) {
    __syncthreads();
    for (int idx = tid; idx < OG * CG; idx += 256) {
      int o = idx / CG, i = idx % CG;
      sW[o * LDH + i] = (_Float16)wm[(g * OG + o) * CG + i];
    }
    if (tid < OG) sBias[tid] = bi[g * OG + tid];
    for (int idx = tid; idx < CG * W_; idx += 256) {       // coalesced over w
      int i = idx / W_, w = idx % W_;
      sX[w * LDH + i] = (_Float16)x[(((size_t)b * CIN + g * CG + i) * H_ + h) * W_ + w];
    }
    __syncthreads();

    // 2 x 7 output tiles of 16x16, K = 128 (4 wmma steps); wave-uniform loop.
    #pragma unroll
    for (int it = 0; it < 2; ++it) {
      int t = wv + 8 * it;
      if (t < 14) {
        int mt = t / 7, nt = t % 7;
        v8f acc = {};
        #pragma unroll
        for (int ks = 0; ks < 4; ++ks) {
          v16h fa = frag_A(sW + mt * 16 * LDH, ks * 32);
          v16h fb = frag_BT(sX + nt * 16 * LDH, ks * 32);
          acc = __builtin_amdgcn_wmma_f32_16x16x32_f16(false, fa, false, fb,
                                                       (short)0, acc, false, false);
        }
        #pragma unroll
        for (int r = 0; r < 8; ++r) {
          int oc = mt * 16 + r + 8 * hh;    // out channel within group
          int w  = nt * 16 + nl;
          outSlab[w * COUT + g * OG + oc] = (_Float16)(acc[r] + sBias[oc]);
        }
      }
    }
  }
}

// ---------------------------------------------------------------------------
// K2: fused score GEMM (Q^T S, 112x112, K=128) + both softmaxes + match + V.
// score lives only in LDS (union re-uses the Q/S staging space after a barrier
// — accumulators ride in registers across it), saving a 90 MB HBM round trip.
// ---------------------------------------------------------------------------
__global__ __launch_bounds__(256) void attn_kernel(
    const _Float16* __restrict__ Qt, const _Float16* __restrict__ St,
    _Float16* __restrict__ Msl, float* __restrict__ matchv,
    float* __restrict__ out) {
  __shared__ __align__(16) union {
    struct { _Float16 q[W_ * LDH]; _Float16 s[W_ * LDH]; } in;           // 60928 B
    struct { float sc[W_ * 113];
             float rmax[W_]; float rinv[W_]; float cmax[W_]; float cinv[W_]; } sm; // 52416 B
  } u;

  const int bh = blockIdx.x;
  const int b = bh / H_, h = bh % H_;
  const int tid = threadIdx.x;
  const int wv = tid >> 5, lane = tid & 31, nl = lane & 15, hh = lane >> 4;

  const size_t slab = (size_t)bh * (W_ * COUT);
  for (int idx = tid; idx < W_ * COUT; idx += 256) {
    int w = idx / COUT, c = idx % COUT;
    u.in.q[w * LDH + c] = Qt[slab + idx];
    u.in.s[w * LDH + c] = St[slab + idx];
  }
  __syncthreads();

  // 7x7 = 49 tiles over 8 waves; keep D tiles in registers.
  v8f acc[7];
  #pragma unroll
  for (int i = 0; i < 7; ++i) {
    int t = wv + 8 * i;
    if (t < 49) {
      int mt = t / 7, nt = t % 7;
      v8f a = {};
      #pragma unroll
      for (int ks = 0; ks < 4; ++ks) {
        v16h fa = frag_A(u.in.q + mt * 16 * LDH, ks * 32);
        v16h fb = frag_BT(u.in.s + nt * 16 * LDH, ks * 32);
        a = __builtin_amdgcn_wmma_f32_16x16x32_f16(false, fa, false, fb,
                                                   (short)0, a, false, false);
      }
      acc[i] = a;
    }
  }
  __syncthreads();                       // Q/S staging no longer needed
  #pragma unroll
  for (int i = 0; i < 7; ++i) {
    int t = wv + 8 * i;
    if (t < 49) {
      int mt = t / 7, nt = t % 7;
      #pragma unroll
      for (int r = 0; r < 8; ++r)
        u.sm.sc[(mt * 16 + r + 8 * hh) * 113 + nt * 16 + nl] = acc[i][r];
    }
  }
  __syncthreads();

  // Row (over v) and column (over w) softmax statistics + match.
  if (tid < W_) {
    int w = tid;
    float mx = -3.0e38f;
    for (int v = 0; v < W_; ++v) mx = fmaxf(mx, u.sm.sc[w * 113 + v]);
    float sme = 0.f, mc = 0.f;
    for (int v = 0; v < W_; ++v) {
      float e = __expf(u.sm.sc[w * 113 + v] - mx);
      sme += e; mc += e * (float)v;
    }
    u.sm.rmax[w] = mx;
    float ri = 1.f / sme;
    u.sm.rinv[w] = ri;
    matchv[bh * W_ + w] = mc * ri;

    int v = tid;
    float cx = -3.0e38f;
    for (int ww = 0; ww < W_; ++ww) cx = fmaxf(cx, u.sm.sc[ww * 113 + v]);
    float cs = 0.f;
    for (int ww = 0; ww < W_; ++ww) cs += __expf(u.sm.sc[ww * 113 + v] - cx);
    u.sm.cmax[v] = cx;
    u.sm.cinv[v] = 1.f / cs;
  }
  __syncthreads();

  // Coalesced write of M_r2l (f16) to workspace.
  for (int idx = tid; idx < W_ * W_; idx += 256) {
    int w = idx / W_, v = idx % W_;
    float e = __expf(u.sm.sc[w * 113 + v] - u.sm.rmax[w]) * u.sm.rinv[w];
    Msl[(size_t)bh * (W_ * W_) + idx] = (_Float16)e;
  }

  // V[w] = tanh(5 * sum_v (sum_{|d|<=2} M_r2l[w+d,v]) * M_l2r[v,w])
  if (tid < W_) {
    int w = tid;
    float vacc = 0.f;
    for (int v = 0; v < W_; ++v) {
      float ec = __expf(u.sm.sc[w * 113 + v] - u.sm.cmax[v]) * u.sm.cinv[v];
      float er = 0.f;
      #pragma unroll
      for (int d = -2; d <= 2; ++d) {
        int wn = w + d;
        if (wn >= 0 && wn < W_)
          er += __expf(u.sm.sc[wn * 113 + v] - u.sm.rmax[wn]) * u.sm.rinv[wn];
      }
      vacc += er * ec;
    }
    float y = 10.f * vacc;                       // tanh(5x) = 1 - 2/(e^{10x/..}+1), x>=0
    float tv = 1.f - 2.f / (__expf(y) + 1.f);
    out[(((size_t)b * OC + 1024) * H_ + h) * W_ + w] = tv;
  }
}

// ---------------------------------------------------------------------------
// K3: warped_att = M_r2l * x_right^T  (per (b,h): 112x512, K=112 padded to 128)
//     + warped_flow (1-D lerp of x_right row h at column match[w]).
// x_right channels processed in 4 chunks of 128 to stay within 64 KB LDS.
// ---------------------------------------------------------------------------
__global__ __launch_bounds__(256) void warp_kernel(
    const _Float16* __restrict__ Msl, const float* __restrict__ matchv,
    const float* __restrict__ xR, float* __restrict__ out) {
  __shared__ __align__(16) _Float16 sM[W_ * LDH];     // A:  [w][v], v padded w/ zeros
  __shared__ __align__(16) _Float16 sX[128 * LDH];    // B^T:[c][v], v padded w/ zeros

  const int bh = blockIdx.x;
  const int b = bh / H_, h = bh % H_;
  const int tid = threadIdx.x;
  const int wv = tid >> 5, lane = tid & 31, nl = lane & 15, hh = lane >> 4;

  for (int idx = tid; idx < W_ * LDH; idx += 256) {
    int w = idx / LDH, v = idx % LDH;
    sM[idx] = (v < W_) ? Msl[(size_t)bh * (W_ * W_) + w * W_ + v] : (_Float16)0.f;
  }
  __syncthreads();

  for (int cc = 0; cc < 4; ++cc) {
    for (int idx = tid; idx < 128 * LDH; idx += 256) {
      int cl = idx / LDH, v = idx % LDH;
      sX[idx] = (v < W_)
          ? (_Float16)xR[(((size_t)b * CIN + cc * 128 + cl) * H_ + h) * W_ + v]
          : (_Float16)0.f;
    }
    __syncthreads();

    // 7 (w) x 8 (c) tiles = 56 tiles over 8 waves, 7 per wave; K = 128 (16 zero-pad).
    #pragma unroll
    for (int i = 0; i < 7; ++i) {
      int t = wv + 8 * i;
      int mt = t / 8, nt = t % 8;
      v8f acc = {};
      #pragma unroll
      for (int ks = 0; ks < 4; ++ks) {
        v16h fa = frag_A(sM + mt * 16 * LDH, ks * 32);
        v16h fb = frag_BT(sX + nt * 16 * LDH, ks * 32);
        acc = __builtin_amdgcn_wmma_f32_16x16x32_f16(false, fa, false, fb,
                                                     (short)0, acc, false, false);
      }
      #pragma unroll
      for (int r = 0; r < 8; ++r) {
        int w = mt * 16 + r + 8 * hh;
        int c = cc * 128 + nt * 16 + nl;
        out[(((size_t)b * OC + c) * H_ + h) * W_ + w] = acc[r];
      }
    }

    // warped_flow for this channel chunk (f32 gathers straight from HBM/L2).
    for (int idx = tid; idx < 128 * W_; idx += 256) {
      int cl = idx / W_, w = idx % W_;
      int c = cc * 128 + cl;
      float mpos = matchv[bh * W_ + w];
      float fl = floorf(mpos);
      int i0 = min(max((int)fl, 0), W_ - 1);
      int i1 = min(max((int)fl + 1, 0), W_ - 1);
      float fr = mpos - fl;
      size_t base = (((size_t)b * CIN + c) * H_ + h) * W_;
      float val = (1.f - fr) * xR[base + i0] + fr * xR[base + i1];
      out[(((size_t)b * OC + 512 + c) * H_ + h) * W_ + w] = val;
    }
    __syncthreads();
  }
}

// ---------------------------------------------------------------------------
extern "C" void kernel_launch(void* const* d_in, const int* in_sizes, int n_in,
                              void* d_out, int out_size, void* d_ws, size_t ws_size,
                              hipStream_t stream) {
  const float* xL  = (const float*)d_in[0];
  const float* xR  = (const float*)d_in[1];
  const float* bqw = (const float*)d_in[2];
  const float* bqb = (const float*)d_in[3];
  const float* bsw = (const float*)d_in[4];
  const float* bsb = (const float*)d_in[5];
  float* out = (float*)d_out;

  // Workspace: Qt + St (f16 [bh][w][c]) = 51.4 MB, M_r2l (f16) = 22.5 MB,
  // match (f32) = 0.4 MB.  Total ~74.3 MB.
  _Float16* Qt  = (_Float16*)d_ws;
  _Float16* St  = Qt + (size_t)NBH * W_ * COUT;
  _Float16* Msl = St + (size_t)NBH * W_ * COUT;
  float* matchv = (float*)(Msl + (size_t)NBH * W_ * W_);

  gconv_kernel<<<dim3(NBH, 2), 256, 0, stream>>>(xL, xR, bqw, bqb, bsw, bsb, Qt, St);
  attn_kernel<<<NBH, 256, 0, stream>>>(Qt, St, Msl, matchv, out);
  warp_kernel<<<NBH, 256, 0, stream>>>(Msl, matchv, xR, out);
}
